// SelfAttention_42408507081185
// MI455X (gfx1250) — matrix operand order repaired and checked by reading, not compiled, not run on hardware
//
#include <hip/hip_runtime.h>

// ---------------------------------------------------------------------------
// Causal multi-head self-attention for MI455X (gfx1250), bf16 WMMA pipeline.
//  B=2 T=2048 C=1024 H=16 D=64.  All matmuls via v_wmma_f32_16x16x32_bf16.
//  GEMM stages the shared weight tile into LDS with gfx1250 async copies
//  (GLOBAL_LOAD_ASYNC_TO_LDS_B128, ASYNCcnt) when the builtin is available.
// ---------------------------------------------------------------------------

typedef __bf16 bf16;
typedef __attribute__((ext_vector_type(16))) __bf16 v16bf;
typedef __attribute__((ext_vector_type(8)))  __bf16 v8bf;
typedef __attribute__((ext_vector_type(8)))  float  v8f;
typedef __attribute__((ext_vector_type(4)))  int    v4i;

#define B_   2
#define T_   2048
#define C_   1024
#define H_   16
#define D_   64
#define NROW (B_ * T_)   // 4096

// ---- gfx1250 async global->LDS copy (feature-detected) --------------------
#if defined(__has_builtin)
#if __has_builtin(__builtin_amdgcn_global_load_async_to_lds_b128) && \
    __has_builtin(__builtin_amdgcn_s_wait_asynccnt)
#define USE_ASYNC_LDS 1
#endif
#endif

#if defined(USE_ASYNC_LDS)
typedef __attribute__((address_space(1))) v4i g_v4i;   // global int4
typedef __attribute__((address_space(3))) v4i l_v4i;   // LDS int4
__device__ __forceinline__ void async_copy_b128(const bf16* g, bf16* l) {
  __builtin_amdgcn_global_load_async_to_lds_b128((g_v4i*)g, (l_v4i*)l, 0, 0);
}
__device__ __forceinline__ void async_wait0() {
  __builtin_amdgcn_s_wait_asynccnt(0);
}
#else
__device__ __forceinline__ void async_copy_b128(const bf16* g, bf16* l) {
  *(v8bf*)l = *(const v8bf*)g;   // fallback: vmem load + ds store
}
__device__ __forceinline__ void async_wait0() {}
#endif

__device__ __forceinline__ v16bf cat8(v8bf lo, v8bf hi) {
  return __builtin_shufflevector(lo, hi, 0, 1, 2, 3, 4, 5, 6, 7,
                                         8, 9, 10, 11, 12, 13, 14, 15);
}

__device__ __forceinline__ v8f wmma_bf16(v16bf a, v16bf b, v8f c) {
  // (neg_a, A, neg_b, B, c_mod, C, reuse_a, reuse_b)
  return __builtin_amdgcn_wmma_f32_16x16x32_bf16(false, a, false, b,
                                                 (short)0, c, false, false);
}

// A-operand fragment (16x32, 16-bit): lane L -> row L%16; lanes 0-15 hold
// K {0..15}, lanes 16-31 hold K {16..31} (contiguous, matches B loads).
__device__ __forceinline__ v16bf load_a_frag(const bf16* row_ptr, int sel) {
  v8bf lo = *(const v8bf*)(row_ptr + sel * 16);
  v8bf hi = *(const v8bf*)(row_ptr + sel * 16 + 8);
  return cat8(lo, hi);
}

// ---------------------------------------------------------------------------
// fp32 -> bf16 conversion
// ---------------------------------------------------------------------------
__global__ void cvt_f32_bf16(const float* __restrict__ in,
                             bf16* __restrict__ out, int n) {
  int i = blockIdx.x * blockDim.x + threadIdx.x;
  if (i < n) out[i] = (bf16)in[i];
}

// ---------------------------------------------------------------------------
// NT GEMM: out[m, c] = sum_k A[m,k] * W[c,k] + bias[c]
//   A: bf16 [M, K=1024], W: bf16 [1024, 1024] (both K-contiguous).
// Block = 256 threads = 8 wave32 -> 128 rows x 64 cols per workgroup.
// Weight tile (64 cols x 32 k, 4 KiB) staged in double-buffered LDS via
// async global->LDS copies; A rows loaded directly (read once, reg-reused).
// mode 0: dst bf16, head-split [B,H,T,D]              (Q, K projections)
// mode 1: dst bf16, head-split transposed [B,H,D,T]   (V projection)
// mode 2: dst fp32, plain [M, C]                      (output projection)
// ---------------------------------------------------------------------------
__global__ __launch_bounds__(256)
void gemm_nt_bf16(const bf16* __restrict__ A, const bf16* __restrict__ W,
                  const float* __restrict__ bias, void* __restrict__ dst,
                  int mode) {
  __shared__ __align__(128) bf16 ldsB[2][64 * 32];  // 2 x 4 KiB

  const int tid  = threadIdx.x;
  const int lane = tid & 31;
  const int wave = tid >> 5;          // 0..7
  const int ln   = lane & 15;
  const int sel  = lane >> 4;
  const int m0   = blockIdx.x * 128 + wave * 16;  // 8 waves x 16 rows
  const int n0   = blockIdx.y * 64;               // 4 col tiles of 16

  // stage one 64x32 weight tile: 256 threads x 16B = 4 KiB
  const int sc = tid >> 2;            // staged col 0..63
  const int sh = (tid & 3) * 8;       // k-halfs chunk 0,8,16,24
  const bf16* wstage = W + (size_t)(n0 + sc) * C_ + sh;

  v8f acc[4] = {};
  const bf16* ap = A + (size_t)(m0 + ln) * C_;

  async_copy_b128(wstage, &ldsB[0][sc * 32 + sh]);
  async_wait0();
  __syncthreads();

  int buf = 0;
  for (int k0 = 0; k0 < C_; k0 += 32) {
    if (k0 + 32 < C_)  // prefetch next k-tile into the other buffer
      async_copy_b128(wstage + k0 + 32, &ldsB[buf ^ 1][sc * 32 + sh]);

    v16bf a = load_a_frag(ap + k0, sel);

    // load all four B fragments into distinct regs so the ds_loads can be
    // claused and the WMMAs retire with partial dscnt waits
    v16bf b[4];
#pragma unroll
    for (int t = 0; t < 4; ++t)
      b[t] = *(const v16bf*)&ldsB[buf][(t * 16 + ln) * 32 + sel * 16];
#pragma unroll
    for (int t = 0; t < 4; ++t)
      acc[t] = wmma_bf16(a, b[t], acc[t]);

    async_wait0();
    __syncthreads();
    buf ^= 1;
  }

#pragma unroll
  for (int t = 0; t < 4; ++t) {
    const int c   = n0 + t * 16 + ln;
    const float bb = bias[c];
#pragma unroll
    for (int r = 0; r < 8; ++r) {
      const int grow = m0 + r + sel * 8;   // C-layout: row = vgpr + 8*lanehalf
      float val = acc[t][r] + bb;
      if (mode == 2) {
        ((float*)dst)[(size_t)grow * C_ + c] = val;
      } else {
        const int b2 = grow >> 11;         // / T_
        const int tt = grow & (T_ - 1);
        const int h  = c >> 6;             // / D_
        const int d  = c & (D_ - 1);
        size_t idx;
        if (mode == 0) idx = ((size_t)(b2 * H_ + h) * T_ + tt) * D_ + d;
        else           idx = ((size_t)(b2 * H_ + h) * D_ + d) * T_ + tt;
        ((bf16*)dst)[idx] = (bf16)val;
      }
    }
  }
}

// ---------------------------------------------------------------------------
// Flash attention (causal): block = 256 threads = 8 waves; one block per
// (b*h, 128 q-rows); each wave owns 16 q-rows.
// q,k: bf16 [B,H,T,D]; vt: bf16 [B,H,D,T]; y out: bf16 [B,T,C].
// Streams 32 key-rows per iteration, softmax in exp2 domain.
// ---------------------------------------------------------------------------
__global__ __launch_bounds__(256)
void attn_flash(const bf16* __restrict__ q, const bf16* __restrict__ k,
                const bf16* __restrict__ vt, bf16* __restrict__ y) {
  __shared__ __align__(128) bf16 lds[8 * 16 * 32];  // per-wave P transpose buf

  const int lane = threadIdx.x & 31;
  const int wave = threadIdx.x >> 5;        // 0..7
  const int ln   = lane & 15;
  const int sel  = lane >> 4;
  const int bh   = blockIdx.x;              // b*H + h
  const int bidx = bh >> 4;
  const int h    = bh & (H_ - 1);
  const int q0   = blockIdx.y * 128 + wave * 16;

  const bf16* qp = q  + (size_t)bh * T_ * D_;
  const bf16* kp = k  + (size_t)bh * T_ * D_;
  const bf16* vp = vt + (size_t)bh * D_ * T_;
  bf16* lw = lds + wave * 16 * 32;

  // Q fragments for this wave's 16 rows (D=64 -> two K=32 chunks), loop-invariant
  const bf16* qrow = qp + (size_t)(q0 + ln) * D_;
  const v16bf aq0 = load_a_frag(qrow, sel);
  const v16bf aq1 = load_a_frag(qrow + 32, sel);

  const float c2 = 0.03125f * 1.44269504088896340736f;  // 1/sqrt(C) * log2(e)

  float mrun[8], lrun[8];
  v8f o[4] = {};
#pragma unroll
  for (int r = 0; r < 8; ++r) { mrun[r] = -1e30f; lrun[r] = 0.0f; }

  for (int j = 0; j < q0 + 16; j += 32) {
    // --- scores: two 16x16 tiles (key cols j..j+15, j+16..j+31) ---
    v8f s0 = {}, s1 = {};
    {
      const bf16* k0p = kp + (size_t)(j + ln) * D_ + sel * 16;
      const bf16* k1p = kp + (size_t)(j + 16 + ln) * D_ + sel * 16;
      s0 = wmma_bf16(aq0, *(const v16bf*)k0p, s0);
      s0 = wmma_bf16(aq1, *(const v16bf*)(k0p + 32), s0);
      s1 = wmma_bf16(aq0, *(const v16bf*)k1p, s1);
      s1 = wmma_bf16(aq1, *(const v16bf*)(k1p + 32), s1);
    }

    // V fragments are independent of the softmax: issue the global loads now
    // so they are in flight behind the softmax VALU work.
    v16bf vb[4];
#pragma unroll
    for (int t = 0; t < 4; ++t)
      vb[t] = *(const v16bf*)(vp + (size_t)(t * 16 + ln) * T_ + j + sel * 16);

    // --- causal mask + online softmax (exp2 domain) ---
    float p0[8], p1[8];
#pragma unroll
    for (int r = 0; r < 8; ++r) {
      const int grow = q0 + r + sel * 8;
      const int c0 = j + ln, c1 = j + 16 + ln;
      float e0 = (c0 <= grow) ? s0[r] * c2 : -1e30f;
      float e1 = (c1 <= grow) ? s1[r] * c2 : -1e30f;
      float mx = fmaxf(e0, e1);
#pragma unroll
      for (int sm = 1; sm < 16; sm <<= 1) mx = fmaxf(mx, __shfl_xor(mx, sm, 32));
      const float mnew = fmaxf(mrun[r], mx);
      const float resc = exp2f(mrun[r] - mnew);
      mrun[r] = mnew;
      const float pe0 = exp2f(e0 - mnew);
      const float pe1 = exp2f(e1 - mnew);
      float ps = pe0 + pe1;
#pragma unroll
      for (int sm = 1; sm < 16; sm <<= 1) ps += __shfl_xor(ps, sm, 32);
      lrun[r] = lrun[r] * resc + ps;
      p0[r] = pe0; p1[r] = pe1;
#pragma unroll
      for (int t = 0; t < 4; ++t) o[t][r] *= resc;
    }

    // --- C-layout P -> A-operand layout via per-wave LDS bounce ---
#pragma unroll
    for (int r = 0; r < 8; ++r) {
      const int rl = r + sel * 8;
      lw[rl * 32 + ln]      = (bf16)p0[r];
      lw[rl * 32 + 16 + ln] = (bf16)p1[r];
    }
    const v16bf pa = load_a_frag(lw + ln * 32, sel);

    // --- O += P @ V : four d-tiles of 16 ---
#pragma unroll
    for (int t = 0; t < 4; ++t)
      o[t] = wmma_bf16(pa, vb[t], o[t]);
  }

  // --- normalize and emit y as bf16 [B,T,C] for the output projection ---
#pragma unroll
  for (int t = 0; t < 4; ++t) {
#pragma unroll
    for (int r = 0; r < 8; ++r) {
      const int grow = q0 + r + sel * 8;
      const float val = o[t][r] / lrun[r];
      const size_t idx = ((size_t)(bidx * T_ + grow)) * C_ + h * D_ + t * 16 + ln;
      y[idx] = (bf16)val;
    }
  }
}

// ---------------------------------------------------------------------------
extern "C" void kernel_launch(void* const* d_in, const int* in_sizes, int n_in,
                              void* d_out, int out_size, void* d_ws, size_t ws_size,
                              hipStream_t stream) {
  // setup_inputs order: x, Wk, bk, Wq, bq, Wv, bv, Wo, bo
  const float* x  = (const float*)d_in[0];
  const float* Wk = (const float*)d_in[1];
  const float* bk = (const float*)d_in[2];
  const float* Wq = (const float*)d_in[3];
  const float* bq = (const float*)d_in[4];
  const float* Wv = (const float*)d_in[5];
  const float* bv = (const float*)d_in[6];
  const float* Wo = (const float*)d_in[7];
  const float* bo = (const float*)d_in[8];
  float* out = (float*)d_out;

  // workspace carve-out (~48 MiB): bf16 x, 4 bf16 weights, q, k, vT, y
  char* ws = (char*)d_ws;
  size_t off = 0;
  auto alloc = [&](size_t bytes) -> void* {
    void* p = ws + off;
    off += (bytes + 255) & ~(size_t)255;
    return p;
  };
  bf16* xb  = (bf16*)alloc((size_t)NROW * C_ * sizeof(bf16));
  bf16* wqb = (bf16*)alloc((size_t)C_ * C_ * sizeof(bf16));
  bf16* wkb = (bf16*)alloc((size_t)C_ * C_ * sizeof(bf16));
  bf16* wvb = (bf16*)alloc((size_t)C_ * C_ * sizeof(bf16));
  bf16* wob = (bf16*)alloc((size_t)C_ * C_ * sizeof(bf16));
  bf16* qb  = (bf16*)alloc((size_t)B_ * H_ * T_ * D_ * sizeof(bf16));
  bf16* kb  = (bf16*)alloc((size_t)B_ * H_ * T_ * D_ * sizeof(bf16));
  bf16* vtb = (bf16*)alloc((size_t)B_ * H_ * D_ * T_ * sizeof(bf16));
  bf16* yb  = (bf16*)alloc((size_t)NROW * C_ * sizeof(bf16));

  const int nx = NROW * C_;
  const int nw = C_ * C_;
  cvt_f32_bf16<<<(nx + 255) / 256, 256, 0, stream>>>(x, xb, nx);
  cvt_f32_bf16<<<(nw + 255) / 256, 256, 0, stream>>>(Wq, wqb, nw);
  cvt_f32_bf16<<<(nw + 255) / 256, 256, 0, stream>>>(Wk, wkb, nw);
  cvt_f32_bf16<<<(nw + 255) / 256, 256, 0, stream>>>(Wv, wvb, nw);
  cvt_f32_bf16<<<(nw + 255) / 256, 256, 0, stream>>>(Wo, wob, nw);

  dim3 ggrid(NROW / 128, C_ / 64);  // 32 x 16 workgroups, 8 waves each
  gemm_nt_bf16<<<ggrid, 256, 0, stream>>>(xb, wqb, bq, qb, 0);
  gemm_nt_bf16<<<ggrid, 256, 0, stream>>>(xb, wkb, bk, kb, 0);
  gemm_nt_bf16<<<ggrid, 256, 0, stream>>>(xb, wvb, bv, vtb, 1);

  dim3 agrid(B_ * H_, T_ / 128);    // 32 x 16 workgroups, 8 waves each
  attn_flash<<<agrid, 256, 0, stream>>>(qb, kb, vtb, yb);

  gemm_nt_bf16<<<ggrid, 256, 0, stream>>>(yb, wob, bo, out, 2);
}
